// GINNode_14525579395559
// MI455X (gfx1250) — compile-verified, matrix operand order checked by reading, not run on hardware
//
#include <hip/hip_runtime.h>
#include <hip/hip_bf16.h>

#define N_NODES 50000
#define N_EDGES 800000
#define DFEAT   128
#define N_LAYERS 3
#define BN_EPS 1e-5f

typedef __attribute__((ext_vector_type(16))) __bf16 v16bf;
typedef __attribute__((ext_vector_type(8)))  float  v8f;

union FragB {
    v16bf v;
    unsigned int u[8];
};

// ---------------------------------------------------------------------------
// zero fill (grid-stride)
// ---------------------------------------------------------------------------
__global__ void zero_kernel(float* __restrict__ p, int n) {
    int i = blockIdx.x * blockDim.x + threadIdx.x;
    int stride = gridDim.x * blockDim.x;
    for (; i < n; i += stride) p[i] = 0.0f;
}

// ---------------------------------------------------------------------------
// GIN aggregation: agg[dst] += h[src], one wave per edge, float4 per lane
// 32 lanes * 4 floats = 128-dim row. L2-resident float atomics.
// ---------------------------------------------------------------------------
__global__ void scatter_add_kernel(const float* __restrict__ h,
                                   const long long* __restrict__ src,
                                   const long long* __restrict__ dst,
                                   float* __restrict__ agg, int nE) {
    int e = blockIdx.x * (blockDim.x >> 5) + (threadIdx.x >> 5);
    if (e >= nE) return;
    int lane = threadIdx.x & 31;
    long long s = src[e];
    long long d = dst[e];
    const float4 v = *((const float4*)(h + (size_t)s * DFEAT) + lane);
    float* pd = agg + (size_t)d * DFEAT + lane * 4;
    unsafeAtomicAdd(pd + 0, v.x);
    unsafeAtomicAdd(pd + 1, v.y);
    unsafeAtomicAdd(pd + 2, v.z);
    unsafeAtomicAdd(pd + 3, v.w);
}

// ---------------------------------------------------------------------------
// Pack a K=128 x Ncols row-major f32 weight matrix into per-lane WMMA
// B-fragment order (bf16):  index = (((ks*NT + tn)*32 + lane)*16 + i)
//   lane: n = tn*16 + (lane&15), khalf = lane>>4
//   i<8 : k = ks*32 + khalf*8 + i
//   i>=8: k = ks*32 + 16 + khalf*8 + (i-8)
// Columns >= Ncols are zero-padded (used for the 40-col output projection).
// ---------------------------------------------------------------------------
__global__ void pack_b_kernel(const float* __restrict__ W,
                              __bf16* __restrict__ out,
                              int Ncols, int NT) {
    int id = blockIdx.x * blockDim.x + threadIdx.x;
    int total = 4 * NT * 32 * 16;
    if (id >= total) return;
    int i    = id & 15;
    int lane = (id >> 4) & 31;
    int tn   = (id >> 9) % NT;
    int ks   = id / (512 * NT);
    int khalf = lane >> 4;
    int nloc  = lane & 15;
    int k = ks * 32 + khalf * 8 + ((i < 8) ? i : (16 + (i - 8)));
    int n = tn * 16 + nloc;
    float w = (n < Ncols) ? W[(size_t)k * Ncols + n] : 0.0f;
    out[id] = (__bf16)w;   // native v_cvt (RNE)
}

// ---------------------------------------------------------------------------
// WMMA GEMM:  Out[M x NOUT] = op(A [+ A2]) @ Bpacked + bias
//   - ADD2   : A := A + A2 elementwise (GIN self + aggregate)
//   - AFF    : A := relu(affA[k]*A + affC[k])  (fused BatchNorm+ReLU on input)
//   - RELUO  : relu on output
//   - STATS  : accumulate per-column sum / sumsq of the output (for BN)
// One wave per 16x16 output tile; block = NT waves covering NT column tiles.
// K = 128 -> 4 x v_wmma_f32_16x16x32_bf16 with f32 accumulation.
// ---------------------------------------------------------------------------
template <int ADD2, int AFF, int RELUO, int STATS, int NT, int NOUT>
__global__ void gemm_wmma_kernel(const float* __restrict__ A,
                                 const float* __restrict__ A2,
                                 const float* __restrict__ affA,
                                 const float* __restrict__ affC,
                                 const __bf16* __restrict__ Bp,
                                 const float* __restrict__ bias,
                                 float* __restrict__ Out,
                                 float* __restrict__ statSum,
                                 float* __restrict__ statSq,
                                 int M) {
    const int tileM = blockIdx.x;
    const int wave  = threadIdx.x >> 5;   // column tile
    const int lane  = threadIdx.x & 31;
    const int khalf = lane >> 4;
    const int row   = tileM * 16 + (lane & 15);

    v8f acc = {};

#pragma unroll
    for (int ks = 0; ks < 4; ++ks) {
        const int kbase = ks * 32 + khalf * 8;

        // ---- A fragment: two contiguous 8-float runs per lane ----
        const float4* pa = (const float4*)(A + (size_t)row * DFEAT + kbase);
        float4 f0 = pa[0], f1 = pa[1];        // k .. k+7
        float4 f2 = pa[4], f3 = pa[5];        // k+16 .. k+23
        if (ADD2) {
            const float4* p2 = (const float4*)(A2 + (size_t)row * DFEAT + kbase);
            float4 g0 = p2[0], g1 = p2[1], g2 = p2[4], g3 = p2[5];
            f0.x += g0.x; f0.y += g0.y; f0.z += g0.z; f0.w += g0.w;
            f1.x += g1.x; f1.y += g1.y; f1.z += g1.z; f1.w += g1.w;
            f2.x += g2.x; f2.y += g2.y; f2.z += g2.z; f2.w += g2.w;
            f3.x += g3.x; f3.y += g3.y; f3.z += g3.z; f3.w += g3.w;
        }
        float fv[16] = {f0.x, f0.y, f0.z, f0.w, f1.x, f1.y, f1.z, f1.w,
                        f2.x, f2.y, f2.z, f2.w, f3.x, f3.y, f3.z, f3.w};
        if (AFF) {
#pragma unroll
            for (int j = 0; j < 16; ++j) {
                int k = kbase + ((j < 8) ? j : (8 + j));  // j>=8 -> kbase+16+(j-8)
                float t = fmaf(affA[k], fv[j], affC[k]);
                fv[j] = fmaxf(t, 0.0f);
            }
        }
        // native f32->bf16 converts (v_cvt_pk_bf16_f32) instead of bit twiddling
        v16bf av;
#pragma unroll
        for (int j = 0; j < 16; ++j) av[j] = (__bf16)fv[j];

        // ---- B fragment: 32 contiguous bytes per lane (2 x b128) ----
        FragB b;
        const uint4* pb = (const uint4*)Bp + ((size_t)(ks * NT + wave) * 32 + lane) * 2;
        uint4 q0 = pb[0], q1 = pb[1];
        b.u[0] = q0.x; b.u[1] = q0.y; b.u[2] = q0.z; b.u[3] = q0.w;
        b.u[4] = q1.x; b.u[5] = q1.y; b.u[6] = q1.z; b.u[7] = q1.w;

        acc = __builtin_amdgcn_wmma_f32_16x16x32_bf16(
            false, av, false, b.v, (short)0, acc, false, false);
    }

    // ---- epilogue: D layout -> row = v + 8*khalf, col = lane&15 ----
    const int nloc = lane & 15;
    const int col  = wave * 16 + nloc;
    const float bv = (col < NOUT) ? bias[col] : 0.0f;
    float s8 = 0.0f, ss8 = 0.0f;
#pragma unroll
    for (int v = 0; v < 8; ++v) {
        int r = tileM * 16 + v + 8 * khalf;
        float val = acc[v] + bv;
        if (RELUO) val = fmaxf(val, 0.0f);
        if (r < M && col < NOUT) Out[(size_t)r * NOUT + col] = val;
        if (STATS) { s8 += val; ss8 += val * val; }
    }
    if (STATS) {
        // lanes L and L^16 hold the same column -> pairwise reduce, one atomic
        s8  += __shfl_xor(s8, 16, 32);
        ss8 += __shfl_xor(ss8, 16, 32);
        if (khalf == 0) {
            unsafeAtomicAdd(&statSum[col], s8);
            unsafeAtomicAdd(&statSq[col], ss8);
        }
    }
}

// ---------------------------------------------------------------------------
// BatchNorm finalize: a = gamma*rsqrt(var+eps), c = beta - mean*a
// ---------------------------------------------------------------------------
__global__ void bn_finalize_kernel(const float* __restrict__ sum,
                                   const float* __restrict__ sq,
                                   const float* __restrict__ gamma,
                                   const float* __restrict__ beta,
                                   float* __restrict__ a,
                                   float* __restrict__ c, float invN) {
    int i = threadIdx.x;
    float mu  = sum[i] * invN;
    float var = sq[i] * invN - mu * mu;
    float s   = gamma[i] * rsqrtf(var + BN_EPS);
    a[i] = s;
    c[i] = beta[i] - mu * s;
}

// ---------------------------------------------------------------------------
// Row-wise log_softmax over 40 logits
// ---------------------------------------------------------------------------
__global__ void log_softmax_kernel(const float* __restrict__ logits,
                                   float* __restrict__ out, int M) {
    int r = blockIdx.x * blockDim.x + threadIdx.x;
    if (r >= M) return;
    const float* p = logits + (size_t)r * 40;
    float mx = -INFINITY;
#pragma unroll
    for (int i = 0; i < 40; ++i) mx = fmaxf(mx, p[i]);
    float s = 0.0f;
#pragma unroll
    for (int i = 0; i < 40; ++i) s += __expf(p[i] - mx);
    float ls = __logf(s);
    float* o = out + (size_t)r * 40;
#pragma unroll
    for (int i = 0; i < 40; ++i) o[i] = p[i] - mx - ls;
}

// ---------------------------------------------------------------------------
// launch
// ---------------------------------------------------------------------------
extern "C" void kernel_launch(void* const* d_in, const int* in_sizes, int n_in,
                              void* d_out, int out_size, void* d_ws, size_t ws_size,
                              hipStream_t stream) {
    const float*     x     = (const float*)d_in[0];
    const long long* edges = (const long long*)d_in[1];   // int64 [2, N_EDGES]
    const float* cW1   = (const float*)d_in[2];
    const float* cb1   = (const float*)d_in[3];
    const float* gamma = (const float*)d_in[4];
    const float* beta  = (const float*)d_in[5];
    const float* cW2   = (const float*)d_in[6];
    const float* cb2   = (const float*)d_in[7];
    const float* mW1   = (const float*)d_in[8];
    const float* mb1   = (const float*)d_in[9];
    const float* mW2   = (const float*)d_in[10];
    const float* mb2   = (const float*)d_in[11];
    float* out = (float*)d_out;

    const size_t NF = (size_t)N_NODES * DFEAT;            // 6,400,000 floats
    float* ws    = (float*)d_ws;
    float* h     = ws;
    float* agg   = ws + NF;
    float* z     = ws + 2 * NF;
    float* stats = ws + 3 * NF;          // [sum(128) | sq(128) | affA(128) | affC(128)]
    __bf16* packs = (__bf16*)(ws + 3 * NF + 512);
    __bf16* pW1 = packs;                 // 3 * 16384
    __bf16* pW2 = packs + 3 * 16384;     // 3 * 16384
    __bf16* pM1 = packs + 6 * 16384;     // 16384
    __bf16* pM2 = packs + 7 * 16384;     // 6144 (40 cols -> 3 tiles, padded)

    const long long* src = edges;
    const long long* dst = edges + N_EDGES;

    // h <- x
    hipMemcpyAsync(h, x, NF * sizeof(float), hipMemcpyDeviceToDevice, stream);

    // pack all weights to bf16 WMMA B-fragment layout (cheap, deterministic)
    for (int l = 0; l < N_LAYERS; ++l) {
        pack_b_kernel<<<64, 256, 0, stream>>>(cW1 + (size_t)l * 16384, pW1 + (size_t)l * 16384, 128, 8);
        pack_b_kernel<<<64, 256, 0, stream>>>(cW2 + (size_t)l * 16384, pW2 + (size_t)l * 16384, 128, 8);
    }
    pack_b_kernel<<<64, 256, 0, stream>>>(mW1, pM1, 128, 8);
    pack_b_kernel<<<24, 256, 0, stream>>>(mW2, pM2, 40, 3);

    const int rowTiles = N_NODES / 16;          // 3125 (exact)
    for (int l = 0; l < N_LAYERS; ++l) {
        zero_kernel<<<4096, 256, 0, stream>>>(agg, (int)NF);
        zero_kernel<<<1, 256, 0, stream>>>(stats, 256);
        scatter_add_kernel<<<N_EDGES / 8, 256, 0, stream>>>(h, src, dst, agg, N_EDGES);

        // z = (h + agg) @ W1 + b1   (+ column stats for BN)
        gemm_wmma_kernel<1, 0, 0, 1, 8, 128><<<rowTiles, 256, 0, stream>>>(
            h, agg, nullptr, nullptr, pW1 + (size_t)l * 16384, cb1 + l * 128,
            z, stats, stats + 128, N_NODES);

        bn_finalize_kernel<<<1, 128, 0, stream>>>(
            stats, stats + 128, gamma + l * 128, beta + l * 128,
            stats + 256, stats + 384, 1.0f / N_NODES);

        // h = relu( relu(BN(z)) @ W2 + b2 )
        gemm_wmma_kernel<0, 1, 1, 0, 8, 128><<<rowTiles, 256, 0, stream>>>(
            z, nullptr, stats + 256, stats + 384, pW2 + (size_t)l * 16384,
            cb2 + l * 128, h, nullptr, nullptr, N_NODES);
    }

    // t = relu(h @ mlp_W1 + b1)   (reuse agg buffer)
    float* t = agg;
    gemm_wmma_kernel<0, 0, 1, 0, 8, 128><<<rowTiles, 256, 0, stream>>>(
        h, nullptr, nullptr, nullptr, pM1, mb1, t, nullptr, nullptr, N_NODES);

    // logits = t @ mlp_W2 + b2   (40 cols -> 3 column tiles, masked stores)
    float* logits = z;
    gemm_wmma_kernel<0, 0, 0, 0, 3, 40><<<rowTiles, 96, 0, stream>>>(
        t, nullptr, nullptr, nullptr, pM2, mb2, logits, nullptr, nullptr, N_NODES);

    log_softmax_kernel<<<(N_NODES + 255) / 256, 256, 0, stream>>>(logits, out, N_NODES);
}
